// WeightGCN_65214783423069
// MI455X (gfx1250) — matrix operand order repaired and checked by reading, not compiled, not run on hardware
//
#include <hip/hip_runtime.h>

// WeightGCN for MI455X (gfx1250, wave32, WMMA).
//
// Strategy:
//  * G@G and G2@G (2 x 1.1 TFLOP) dominate -> run them on
//    V_WMMA_F32_16X16X32_BF16 using a bf16 hi/lo error-compensated split
//    (3 WMMA products per tile, ~f32 accuracy, ~2.7x faster than the K=4
//    native f32 WMMA path). f32 accumulate.
//  * Double-buffered LDS tiles: next K-slab is fetched while the current
//    one feeds the WMMA pipe; A tiles use the CDNA5 async global->LDS DMA
//    path (GLOBAL_LOAD_ASYNC_TO_LDS_B128, ASYNCcnt) so they never touch
//    VGPRs; B tiles go through VGPRs because they are transposed into LDS.
//  * Row softmax stats (masked max / exp-sum) in a single memory pass.
//  * Aggregation P @ embed fuses the masked softmax into the WMMA GEMM's
//    A-tile producer (exp applied in registers, split to bf16 hi/lo into
//    LDS) -> probability matrices are never materialized.
//
// Workspace layout (needs ~772 MB):
//   Ghi,Glo,G2hi,G2lo,G3hi,G3lo : 6 * 8192^2 * 2B = 768 MB
//   embed hi/lo, per-layer row max / inv-sum : ~4.4 MB

typedef __attribute__((ext_vector_type(16))) __bf16 v16bf;
typedef __attribute__((ext_vector_type(8)))  float  v8f;

#define TN 8192   // graph dimension
#define TD 128    // embed dimension
#define BM 128
#define BN 128
#define BK 32
#define LDT 40    // padded LDS row stride (elements): 80B rows -> conflict-free b128 frag loads

#define TILE_ELEMS (BM * LDT)                    // 5120 bf16 per matrix tile
#define SMEM_BYTES (8 * TILE_ELEMS * 2)          // 2 stages * (Ahi,Alo,Bhi,Blo) = 80 KB

union BVec { uint4 u; __bf16 b[8]; };
union Frag { v16bf v; uint4 u[2]; };

// A fragment (16x32 bf16): lane L (<16): row L, K[0:8) and K[16:24);
// lane L (>=16): row L-16, K[8:16) and K[24:32).
__device__ __forceinline__ v16bf frag_a(const __bf16* base, int lane) {
  const __bf16* p = base + (lane & 15) * LDT + ((lane >> 4) << 3);
  Frag f;
  f.u[0] = *reinterpret_cast<const uint4*>(p);
  f.u[1] = *reinterpret_cast<const uint4*>(p + 16);
  return f.v;
}
// B fragment (32x16 bf16), B stored transposed in LDS as [n][k]:
// lane L (<16): col L, K[0:8) and K[16:24); lane L (>=16): col L-16, K[8:16) and K[24:32).
__device__ __forceinline__ v16bf frag_b(const __bf16* base, int lane) {
  const __bf16* p = base + (lane & 15) * LDT + ((lane >> 4) << 3);
  Frag f;
  f.u[0] = *reinterpret_cast<const uint4*>(p);
  f.u[1] = *reinterpret_cast<const uint4*>(p + 16);
  return f.v;
}

// Error-compensated product: (ah+al) @ (bh+bl) ~= ah@bh + ah@bl + al@bh
__device__ __forceinline__ v8f wmma3(v16bf ah, v16bf al, v16bf bh, v16bf bl, v8f c) {
  c = __builtin_amdgcn_wmma_f32_16x16x32_bf16(false, ah, false, bh, (short)0, c, false, false);
  c = __builtin_amdgcn_wmma_f32_16x16x32_bf16(false, ah, false, bl, (short)0, c, false, false);
  c = __builtin_amdgcn_wmma_f32_16x16x32_bf16(false, al, false, bh, (short)0, c, false, false);
  return c;
}

// Shared WMMA core: wave computes a 64x32 patch (4x2 frags) of the 128x128 block.
__device__ __forceinline__ void mma_tiles(const __bf16* sA0, const __bf16* sA1,
                                          const __bf16* sB0, const __bf16* sB1,
                                          int wm, int wn, int lane, v8f (&acc)[4][2]) {
  v16bf a[4][2], b[2][2];
#pragma unroll
  for (int mi = 0; mi < 4; ++mi) {
    a[mi][0] = frag_a(sA0 + (wm + mi * 16) * LDT, lane);
    a[mi][1] = frag_a(sA1 + (wm + mi * 16) * LDT, lane);
  }
#pragma unroll
  for (int ni = 0; ni < 2; ++ni) {
    b[ni][0] = frag_b(sB0 + (wn + ni * 16) * LDT, lane);
    b[ni][1] = frag_b(sB1 + (wn + ni * 16) * LDT, lane);
  }
#pragma unroll
  for (int mi = 0; mi < 4; ++mi)
#pragma unroll
    for (int ni = 0; ni < 2; ++ni)
      acc[mi][ni] = wmma3(a[mi][0], a[mi][1], b[ni][0], b[ni][1], acc[mi][ni]);
}

__device__ __forceinline__ unsigned lds_u32(const void* p) {
  // LDS aperture: generic address low 32 bits are the LDS byte address.
  return (unsigned)(unsigned long long)(uintptr_t)p;
}
__device__ __forceinline__ void async_copy_b128(const void* gsrc, void* ldst) {
  unsigned la = lds_u32(ldst);
  unsigned long long ga = (unsigned long long)(uintptr_t)gsrc;
  asm volatile("global_load_async_to_lds_b128 %0, %1, off" :: "v"(la), "v"(ga) : "memory");
}
__device__ __forceinline__ void wait_async0() {
  asm volatile("s_wait_asynccnt 0x0" ::: "memory");
}

// ---------------------------------------------------------------- splits ----
__global__ void __launch_bounds__(256)
split_bf16(const float* __restrict__ X, __bf16* __restrict__ Hi, __bf16* __restrict__ Lo) {
  size_t v = (size_t)blockIdx.x * 256 + threadIdx.x;  // one 8-element vector
  const float4* xp = reinterpret_cast<const float4*>(X) + v * 2;
  float4 a = xp[0], b = xp[1];
  float xs[8] = {a.x, a.y, a.z, a.w, b.x, b.y, b.z, b.w};
  BVec h, l;
#pragma unroll
  for (int j = 0; j < 8; ++j) {
    __bf16 hb = (__bf16)xs[j];
    h.b[j] = hb;
    l.b[j] = (__bf16)(xs[j] - (float)hb);
  }
  reinterpret_cast<uint4*>(Hi)[v] = h.u;
  reinterpret_cast<uint4*>(Lo)[v] = l.u;
}

// ------------------------------------------------------ graph-power GEMM ----
__global__ void __launch_bounds__(256)
pow_gemm(const __bf16* __restrict__ Ahi, const __bf16* __restrict__ Alo,
         const __bf16* __restrict__ Bhi, const __bf16* __restrict__ Blo,
         __bf16* __restrict__ Chi, __bf16* __restrict__ Clo) {
  extern __shared__ __bf16 smem[];  // [A,B][2 stages][hi,lo]
  const int tid = threadIdx.x, lane = tid & 31, wave = tid >> 5;
  const int wm = (wave >> 2) * 64, wn = (wave & 3) * 32;
  const int bm = blockIdx.y * BM, bn = blockIdx.x * BN;

  auto sA = [&](int st, int h) { return smem + (st * 2 + h) * TILE_ELEMS; };
  auto sB = [&](int st, int h) { return smem + 4 * TILE_ELEMS + (st * 2 + h) * TILE_ELEMS; };

  auto load_tiles = [&](int st, int kt) {
    // A tiles 128x32: async DMA straight into LDS (16B per lane, no VGPRs).
#pragma unroll
    for (int j = 0; j < 2; ++j) {
      int v = (wave * 2 + j) * 32 + lane;  // 512 16B vectors over 8 waves
      int r = v >> 2, kc = (v & 3) * 8;
      size_t g = (size_t)(bm + r) * TN + kt + kc;
      async_copy_b128(Ahi + g, sA(st, 0) + r * LDT + kc);
      async_copy_b128(Alo + g, sA(st, 1) + r * LDT + kc);
    }
    // B tile 32x128, transposed into LDS as [n][k] (through VGPRs).
#pragma unroll
    for (int i = 0; i < 2; ++i) {
      int v = tid + i * 256;
      int kr = v >> 4, nc = (v & 15) * 8;
      size_t g = (size_t)(kt + kr) * TN + bn + nc;
      BVec th, tl;
      th.u = *reinterpret_cast<const uint4*>(Bhi + g);
      tl.u = *reinterpret_cast<const uint4*>(Blo + g);
      __bf16* b0 = sB(st, 0);
      __bf16* b1 = sB(st, 1);
#pragma unroll
      for (int j = 0; j < 8; ++j) {
        b0[(nc + j) * LDT + kr] = th.b[j];
        b1[(nc + j) * LDT + kr] = tl.b[j];
      }
    }
  };

  const v8f vz = {0.f, 0.f, 0.f, 0.f, 0.f, 0.f, 0.f, 0.f};
  v8f acc[4][2];
#pragma unroll
  for (int mi = 0; mi < 4; ++mi)
#pragma unroll
    for (int ni = 0; ni < 2; ++ni) acc[mi][ni] = vz;

  load_tiles(0, 0);
  wait_async0();
  __syncthreads();

  for (int kt = 0; kt < TN; kt += BK) {
    int cur = (kt >> 5) & 1;
    if (kt + BK < TN) load_tiles(cur ^ 1, kt + BK);     // prefetch next stage
    mma_tiles(sA(cur, 0), sA(cur, 1), sB(cur, 0), sB(cur, 1), wm, wn, lane, acc);
    wait_async0();   // our async writes to stage cur^1 are visible
    __syncthreads(); // everyone done reading cur, done writing cur^1
  }

  // Epilogue: store f32 accumulators directly as bf16 hi/lo (exact-zero safe).
#pragma unroll
  for (int mi = 0; mi < 4; ++mi)
#pragma unroll
    for (int ni = 0; ni < 2; ++ni)
#pragma unroll
      for (int e = 0; e < 8; ++e) {
        int row = bm + wm + mi * 16 + e + ((lane >> 4) << 3);
        int col = bn + wn + ni * 16 + (lane & 15);
        float x = acc[mi][ni][e];
        __bf16 h = (__bf16)x;
        size_t g = (size_t)row * TN + col;
        Chi[g] = h;
        Clo[g] = (__bf16)(x - (float)h);
      }
}

// --------------------------------------------------- masked softmax stats ---
__global__ void __launch_bounds__(256)
row_stats(const float* __restrict__ Gf, const __bf16* __restrict__ Ghi,
          const __bf16* __restrict__ Glo, float* __restrict__ mrow,
          float* __restrict__ isrow, int use_f32) {
  const int row = blockIdx.x, tid = threadIdx.x;
  float x[32];
  if (use_f32) {
    const float* g = Gf + (size_t)row * TN;
#pragma unroll
    for (int i = 0; i < 32; ++i) x[i] = g[tid + i * 256];
  } else {
    const __bf16* gh = Ghi + (size_t)row * TN;
    const __bf16* gl = Glo + (size_t)row * TN;
#pragma unroll
    for (int i = 0; i < 32; ++i)
      x[i] = (float)gh[tid + i * 256] + (float)gl[tid + i * 256];
  }
  float m = -1e30f;
#pragma unroll
  for (int i = 0; i < 32; ++i)
    if (x[i] != 0.f) m = fmaxf(m, x[i]);

  __shared__ float red[256];
  red[tid] = m;
  __syncthreads();
  for (int s = 128; s > 0; s >>= 1) {
    if (tid < s) red[tid] = fmaxf(red[tid], red[tid + s]);
    __syncthreads();
  }
  m = red[0];
  float safe_m = (m > -5e29f) ? m : 0.f;
  __syncthreads();

  float sum = 0.f;
#pragma unroll
  for (int i = 0; i < 32; ++i)
    if (x[i] != 0.f) sum += __expf(x[i] - safe_m);
  red[tid] = sum;
  __syncthreads();
  for (int s = 128; s > 0; s >>= 1) {
    if (tid < s) red[tid] += red[tid + s];
    __syncthreads();
  }
  if (tid == 0) {
    mrow[row] = safe_m;
    isrow[row] = 1.f / fmaxf(red[0], 1e-30f);
  }
}

// ----------------------------- fused softmax(P) @ embed aggregation GEMM ----
__global__ void __launch_bounds__(256)
agg_gemm(const float* __restrict__ Af, const __bf16* __restrict__ Ahi,
         const __bf16* __restrict__ Alo, const float* __restrict__ mrow,
         const float* __restrict__ isrow, const __bf16* __restrict__ Ehi,
         const __bf16* __restrict__ Elo, float* __restrict__ embs,
         int use_f32, int slot) {
  extern __shared__ __bf16 smem[];
  const int tid = threadIdx.x, lane = tid & 31, wave = tid >> 5;
  const int wm = (wave >> 2) * 64, wn = (wave & 3) * 32;
  const int bm = blockIdx.x * BM;  // bn == 0 (D = 128 == BN)

  auto sA = [&](int st, int h) { return smem + (st * 2 + h) * TILE_ELEMS; };
  auto sB = [&](int st, int h) { return smem + 4 * TILE_ELEMS + (st * 2 + h) * TILE_ELEMS; };

  auto load_tiles = [&](int st, int kt) {
    // A tile: load graph-power values, apply masked softmax, split to hi/lo.
#pragma unroll
    for (int i = 0; i < 2; ++i) {
      int v = tid + i * 256;
      int r = v >> 2, kc = (v & 3) * 8;
      float xs[8];
      if (use_f32) {
        const float4* gp = reinterpret_cast<const float4*>(Af + (size_t)(bm + r) * TN + kt + kc);
        float4 p0 = gp[0], p1 = gp[1];
        xs[0] = p0.x; xs[1] = p0.y; xs[2] = p0.z; xs[3] = p0.w;
        xs[4] = p1.x; xs[5] = p1.y; xs[6] = p1.z; xs[7] = p1.w;
      } else {
        size_t g = (size_t)(bm + r) * TN + kt + kc;
        BVec h, l;
        h.u = *reinterpret_cast<const uint4*>(Ahi + g);
        l.u = *reinterpret_cast<const uint4*>(Alo + g);
#pragma unroll
        for (int j = 0; j < 8; ++j) xs[j] = (float)h.b[j] + (float)l.b[j];
      }
      float mr = mrow[bm + r], is = isrow[bm + r];
      BVec oh, ol;
#pragma unroll
      for (int j = 0; j < 8; ++j) {
        float p = (xs[j] != 0.f) ? __expf(xs[j] - mr) * is : 0.f;
        __bf16 ph = (__bf16)p;
        oh.b[j] = ph;
        ol.b[j] = (__bf16)(p - (float)ph);
      }
      *reinterpret_cast<uint4*>(sA(st, 0) + r * LDT + kc) = oh.u;
      *reinterpret_cast<uint4*>(sA(st, 1) + r * LDT + kc) = ol.u;
    }
    // B tile: embed rows kt..kt+31, transposed into LDS.
#pragma unroll
    for (int i = 0; i < 2; ++i) {
      int v = tid + i * 256;
      int kr = v >> 4, nc = (v & 15) * 8;
      size_t g = (size_t)(kt + kr) * TD + nc;
      BVec th, tl;
      th.u = *reinterpret_cast<const uint4*>(Ehi + g);
      tl.u = *reinterpret_cast<const uint4*>(Elo + g);
      __bf16* b0 = sB(st, 0);
      __bf16* b1 = sB(st, 1);
#pragma unroll
      for (int j = 0; j < 8; ++j) {
        b0[(nc + j) * LDT + kr] = th.b[j];
        b1[(nc + j) * LDT + kr] = tl.b[j];
      }
    }
  };

  const v8f vz = {0.f, 0.f, 0.f, 0.f, 0.f, 0.f, 0.f, 0.f};
  v8f acc[4][2];
#pragma unroll
  for (int mi = 0; mi < 4; ++mi)
#pragma unroll
    for (int ni = 0; ni < 2; ++ni) acc[mi][ni] = vz;

  load_tiles(0, 0);
  __syncthreads();

  for (int kt = 0; kt < TN; kt += BK) {
    int cur = (kt >> 5) & 1;
    if (kt + BK < TN) load_tiles(cur ^ 1, kt + BK);
    mma_tiles(sA(cur, 0), sA(cur, 1), sB(cur, 0), sB(cur, 1), wm, wn, lane, acc);
    __syncthreads();
  }

  // Write this layer's slice of embs[N,4,D].
#pragma unroll
  for (int mi = 0; mi < 4; ++mi)
#pragma unroll
    for (int ni = 0; ni < 2; ++ni)
#pragma unroll
      for (int e = 0; e < 8; ++e) {
        int row = bm + wm + mi * 16 + e + ((lane >> 4) << 3);
        int col = wn + ni * 16 + (lane & 15);
        embs[((size_t)row * 4 + slot) * TD + col] = acc[mi][ni][e];
      }
}

// ------------------------------------------------------------- finalize -----
__global__ void __launch_bounds__(256)
finalize(const float* __restrict__ embed, float* __restrict__ out) {
  size_t idx = (size_t)blockIdx.x * 256 + threadIdx.x;  // over N*D
  size_t i = idx / TD, d = idx % TD;
  float* embs = out + (size_t)TN * TD;
  float e0 = embed[idx];
  float e1 = embs[(i * 4 + 1) * TD + d];
  float e2 = embs[(i * 4 + 2) * TD + d];
  float e3 = embs[(i * 4 + 3) * TD + d];
  embs[(i * 4 + 0) * TD + d] = e0;
  out[idx] = 0.25f * (e0 + e1 + e2 + e3);
}

// ------------------------------------------------------------ launcher ------
extern "C" void kernel_launch(void* const* d_in, const int* in_sizes, int n_in,
                              void* d_out, int out_size, void* d_ws, size_t ws_size,
                              hipStream_t stream) {
  (void)in_sizes; (void)n_in; (void)out_size; (void)ws_size;
  const float* graph = (const float*)d_in[0];   // [8192, 8192] f32
  const float* embed = (const float*)d_in[1];   // [8192, 128]  f32
  float* out = (float*)d_out;                   // light_out [N,D] ++ embs [N,4,D]

  const size_t NN = (size_t)TN * TN;
  char* w = (char*)d_ws;
  auto take = [&](size_t bytes) { char* p = w; w += (bytes + 255) & ~(size_t)255; return p; };
  __bf16* Ghi  = (__bf16*)take(NN * 2);
  __bf16* Glo  = (__bf16*)take(NN * 2);
  __bf16* G2hi = (__bf16*)take(NN * 2);
  __bf16* G2lo = (__bf16*)take(NN * 2);
  __bf16* G3hi = (__bf16*)take(NN * 2);
  __bf16* G3lo = (__bf16*)take(NN * 2);
  __bf16* Ehi  = (__bf16*)take((size_t)TN * TD * 2);
  __bf16* Elo  = (__bf16*)take((size_t)TN * TD * 2);
  float* m0 = (float*)take(TN * 4); float* s0 = (float*)take(TN * 4);
  float* m1 = (float*)take(TN * 4); float* s1 = (float*)take(TN * 4);
  float* m2 = (float*)take(TN * 4); float* s2 = (float*)take(TN * 4);

  // bf16 hi/lo decompositions of graph and embed
  split_bf16<<<(int)(NN / 2048), 256, 0, stream>>>(graph, Ghi, Glo);
  split_bf16<<<(int)((size_t)TN * TD / 2048), 256, 0, stream>>>(embed, Ehi, Elo);

  // graph powers on WMMA (error-compensated bf16, double-buffered, async A)
  dim3 gg(TN / BN, TN / BM);
  pow_gemm<<<gg, 256, SMEM_BYTES, stream>>>(Ghi, Glo, Ghi, Glo, G2hi, G2lo);
  pow_gemm<<<gg, 256, SMEM_BYTES, stream>>>(G2hi, G2lo, Ghi, Glo, G3hi, G3lo);

  // per-row masked softmax stats
  row_stats<<<TN, 256, 0, stream>>>(graph, nullptr, nullptr, m0, s0, 1);
  row_stats<<<TN, 256, 0, stream>>>(nullptr, G2hi, G2lo, m1, s1, 0);
  row_stats<<<TN, 256, 0, stream>>>(nullptr, G3hi, G3lo, m2, s2, 0);

  // fused softmax @ embed -> embs slots 1..3
  float* embs = out + (size_t)TN * TD;
  agg_gemm<<<TN / BM, 256, SMEM_BYTES, stream>>>(graph, nullptr, nullptr, m0, s0, Ehi, Elo, embs, 1, 1);
  agg_gemm<<<TN / BM, 256, SMEM_BYTES, stream>>>(nullptr, G2hi, G2lo, m1, s1, Ehi, Elo, embs, 0, 2);
  agg_gemm<<<TN / BM, 256, SMEM_BYTES, stream>>>(nullptr, G3hi, G3lo, m2, s2, Ehi, Elo, embs, 0, 3);

  // embs slot 0 = embed; light_out = mean over 4 layers
  finalize<<<(int)((size_t)TN * TD / 256), 256, 0, stream>>>(embed, out);
}
